// MultiScaleWalkSampler_47124381172077
// MI455X (gfx1250) — matrix operand order repaired and checked by reading, not compiled, not run on hardware
//
#include <hip/hip_runtime.h>
#include <hip/hip_bf16.h>

typedef __attribute__((ext_vector_type(16))) _Float16 v16h;
typedef __attribute__((ext_vector_type(8)))  float    v8f;

#define NN   100000
#define DD   64
#define BB   4096
#define WW   10
#define LL   8
#define MM   128
#define TT   64
#define KK   (MM + TT)            // 192
#define TOT  (BB * WW)            // 40960 walkers
#define FSTR 196                  // f32 per feature row; 196 % 64 == 4 -> conflict-free b128 reads
#define WPB  2                    // waves per block
#define BLOCK (WPB * 32)          // 64 threads
#define NEGBIG (-1.0e9f)

#if __has_builtin(__builtin_amdgcn_global_load_async_to_lds_b128) && \
    __has_builtin(__builtin_amdgcn_s_wait_asynccnt)
#define USE_ASYNC 1
typedef int v4i __attribute__((vector_size(16)));
typedef __attribute__((address_space(1))) v4i as1_v4i;   // global int4
typedef __attribute__((address_space(3))) v4i as3_v4i;   // LDS int4
#else
#define USE_ASYNC 0
#endif

__global__ __launch_bounds__(BLOCK) void
walk_kernel(const int* __restrict__ nb_ids, const float* __restrict__ nb_times,
            const int* __restrict__ nb_counts, const int* __restrict__ start_nodes,
            const float* __restrict__ start_times, const float* __restrict__ mem,
            const float* __restrict__ time_w, const float* __restrict__ time_b,
            const float* __restrict__ restart_w, const float* __restrict__ restart_b,
            const float* __restrict__ rand_restart, const float* __restrict__ rand_nb,
            float* __restrict__ out_nodes, float* __restrict__ out_times,
            float* __restrict__ out_probs)
{
    __shared__ __align__(16) float feat[WPB][32][FSTR];  // [0,128)=mem row f32, [128,192)=tenc

    const int lane = threadIdx.x & 31;
    const int wv   = threadIdx.x >> 5;
    const int g    = blockIdx.x * BLOCK + threadIdx.x;   // global walker id (< TOT by grid)
    const int b    = g / WW;                             // batch row

    // ---- A-matrix fragments: restart_w broadcast to all 16 rows -------------
    // A k order used here (contiguous, same as B): vgpr v, half h -> K = 2v+h + (lane>=16 ? 16 : 0)
    const int hi8 = (lane & 16) ? 8 : 0;
    v16h afrag[6];
#pragma unroll
    for (int c = 0; c < 6; ++c) {
#pragma unroll
        for (int v = 0; v < 8; ++v) {
#pragma unroll
            for (int h = 0; h < 2; ++h) {
                int kl = 2 * v + h + 2 * hi8;
                afrag[c][2 * v + h] = (_Float16)restart_w[32 * c + kl];
            }
        }
    }

    const float bias = restart_b[0];
    const int   s_n  = start_nodes[b];
    const float s_t  = start_times[b];
    int   cur_n = s_n;
    float cur_t = s_t;

    out_nodes[g * (LL + 1)] = (float)s_n;
    out_times[g * (LL + 1)] = s_t;

#pragma unroll 1
    for (int step = 0; step < LL; ++step) {
        // ================= Phase A: stage f32 features to LDS ================
#if USE_ASYNC
#pragma unroll 1
        for (int w = 0; w < 32; ++w) {
            int   node = __shfl(cur_n, w, 32);
            float t    = __shfl(cur_t, w, 32);
            float* row = &feat[wv][w][0];
            // one async DMA instruction moves the whole 512B row (16B/lane), ASYNCcnt-tracked
            const float* gsrc = mem + (size_t)node * MM + lane * 4;
            __builtin_amdgcn_global_load_async_to_lds_b128(
                (as1_v4i*)gsrc, (as3_v4i*)(row + lane * 4), 0, 0);
            int j = lane * 2;
            float2 tv;
            tv.x = __cosf(t * time_w[j]     + time_b[j]);
            tv.y = __cosf(t * time_w[j + 1] + time_b[j + 1]);
            *(float2*)(row + MM + j) = tv;
        }
        __builtin_amdgcn_s_wait_asynccnt(0);
#else
#pragma unroll 4
        for (int w = 0; w < 32; ++w) {
            int   node = __shfl(cur_n, w, 32);
            float t    = __shfl(cur_t, w, 32);
            float* row = &feat[wv][w][0];
            float4 mv = ((const float4*)(mem + (size_t)node * MM))[lane];
            *(float4*)(row + lane * 4) = mv;
            int j = lane * 2;
            float2 tv;
            tv.x = __cosf(t * time_w[j]     + time_b[j]);
            tv.y = __cosf(t * time_w[j + 1] + time_b[j + 1]);
            *(float2*)(row + MM + j) = tv;
        }
#endif
        __syncthreads();

        // ================= Phase B: WMMA GEMV (logits) =======================
        // D[m][n] = sum_k w[k] * feat[k][n]  (same for every m); cvt f32->f16 in the feed.
        // unroll 2 keeps 4 WMMAs in flight but caps the live f32 set -> stay under 256 VGPRs
        v8f acc0 = {}; v8f acc1 = {};
        const int wk0 = (lane & 15);
        const int wk1 = 16 + (lane & 15);
        const int kb  = (lane & 16) ? 16 : 0;   // B 32x16: lanes16-31 hold K=16..31
#pragma unroll 2
        for (int c = 0; c < 6; ++c) {
            const float* rp0 = &feat[wv][wk0][32 * c + kb];
            const float* rp1 = &feat[wv][wk1][32 * c + kb];
            v16h b0, b1;
#pragma unroll
            for (int i = 0; i < 16; ++i) { b0[i] = (_Float16)rp0[i]; b1[i] = (_Float16)rp1[i]; }
            acc0 = __builtin_amdgcn_wmma_f32_16x16x32_f16(false, afrag[c], false, b0,
                                                          (short)0, acc0, false, false);
            acc1 = __builtin_amdgcn_wmma_f32_16x16x32_f16(false, afrag[c], false, b1,
                                                          (short)0, acc1, false, false);
        }
        __syncthreads();

        float logit = ((lane & 16) ? acc1[0] : acc0[0]) + bias;
        float p     = __builtin_amdgcn_rcpf(1.0f + __expf(-logit));   // sigmoid

        // ================= Phase C: temporal neighbor sampling ===============
        float r_re = rand_restart[(size_t)step * TOT + g];
        float r_nb = rand_nb[(size_t)step * TOT + g];
        bool  restart = r_re < p;

        int cnt = nb_counts[cur_n];
        const float4* tp = (const float4*)(nb_times + (size_t)cur_n * DD);

        // pass 1: online softmax (running max + rescaled sum), track validity
        float mx = -3.0e38f, sum = 0.0f;
        int anyv = 0;
#pragma unroll 4
        for (int q = 0; q < 16; ++q) {
            float4 tv = tp[q];
            float te[4] = {tv.x, tv.y, tv.z, tv.w};
#pragma unroll
            for (int e = 0; e < 4; ++e) {
                int  slot  = q * 4 + e;
                bool valid = (te[e] < cur_t) && (slot < cnt);
                anyv |= (int)valid;
                float sc = valid ? (te[e] - cur_t) * 10.0f : NEGBIG;  // /TEMP
                float mn = fmaxf(mx, sc);
                sum = sum * __expf(mx - mn) + __expf(sc - mn);
                mx  = mn;
            }
        }
        float inv = __builtin_amdgcn_rcpf(sum);

        // pass 2: idx = clip(count(cdf < r), 0, D-1)  (reloads hit L0/L2)
        float cum = 0.0f;
        int count = 0;
#pragma unroll 4
        for (int q = 0; q < 16; ++q) {
            float4 tv = tp[q];
            float te[4] = {tv.x, tv.y, tv.z, tv.w};
#pragma unroll
            for (int e = 0; e < 4; ++e) {
                int  slot  = q * 4 + e;
                bool valid = (te[e] < cur_t) && (slot < cnt);
                float sc = valid ? (te[e] - cur_t) * 10.0f : NEGBIG;
                cum += __expf(sc - mx) * inv;
                count += (cum < r_nb) ? 1 : 0;
            }
        }
        int idx = min(count, DD - 1);

        int   sel_id = nb_ids[(size_t)cur_n * DD + idx];
        float sel_t  = nb_times[(size_t)cur_n * DD + idx];
        int   step_n = anyv ? sel_id : cur_n;
        float step_t = anyv ? sel_t  : cur_t;
        cur_n = restart ? s_n : step_n;
        cur_t = restart ? s_t : step_t;

        // pull next step's gather targets toward the caches
        __builtin_prefetch(mem + (size_t)cur_n * MM, 0, 3);
        __builtin_prefetch(nb_times + (size_t)cur_n * DD, 0, 3);

        out_nodes[g * (LL + 1) + step + 1] = (float)cur_n;
        out_times[g * (LL + 1) + step + 1] = cur_t;
        out_probs[g * LL + step]           = p;
    }
}

extern "C" void kernel_launch(void* const* d_in, const int* in_sizes, int n_in,
                              void* d_out, int out_size, void* d_ws, size_t ws_size,
                              hipStream_t stream) {
    const int*   nb_ids    = (const int*)  d_in[0];
    const float* nb_times  = (const float*)d_in[1];
    const int*   nb_counts = (const int*)  d_in[2];
    const int*   start_n   = (const int*)  d_in[3];
    const float* start_t   = (const float*)d_in[4];
    const float* mem       = (const float*)d_in[5];
    const float* time_w    = (const float*)d_in[6];
    const float* time_b    = (const float*)d_in[7];
    const float* restart_w = (const float*)d_in[8];
    const float* restart_b = (const float*)d_in[9];
    const float* rand_re   = (const float*)d_in[10];
    const float* rand_nb   = (const float*)d_in[11];

    float* out       = (float*)d_out;
    float* out_nodes = out;                              // [B,W,L+1]
    float* out_times = out + (size_t)TOT * (LL + 1);     // [B,W,L+1]
    float* out_probs = out + (size_t)2 * TOT * (LL + 1); // [B,W,L]

    dim3 grid(TOT / BLOCK);   // 640 blocks * 64 threads = 40960 walkers exactly
    dim3 block(BLOCK);
    walk_kernel<<<grid, block, 0, stream>>>(nb_ids, nb_times, nb_counts, start_n, start_t,
                                            mem, time_w, time_b, restart_w, restart_b,
                                            rand_re, rand_nb,
                                            out_nodes, out_times, out_probs);
}
